// TriangleAttn_42099269436004
// MI455X (gfx1250) — matrix operand order repaired and checked by reading, not compiled, not run on hardware
//
#include <hip/hip_runtime.h>
#include <hip/hip_bf16.h>

typedef __attribute__((ext_vector_type(16))) _Float16 v16h;
typedef __attribute__((ext_vector_type(8)))  float    v8f;

#define NH 4
#define DHD 32
#define NN 256
#define DD 128
#define NROWS 65536          // NN*NN
#define PLD 528              // proj leading dim: 384 qkv | 128 g | 4 bias | 12 pad
#define SCALEV 0.17677669529663687f

// workspace layout (bytes)
static const size_t OFF_NORMED = 0;                                  // 65536*128 f16 = 16 MB
static const size_t OFF_WCAT   = 16777216;                           // 528*128 f16
static const size_t OFF_WOUT   = OFF_WCAT + (size_t)528*128*2;       // 128*128 f16
static const size_t OFF_PROJ   = OFF_WOUT + (size_t)128*128*2;       // 65536*528 f16 = 66 MB
static const size_t OFF_ATTN   = OFF_PROJ + (size_t)NROWS*PLD*2;     // 65536*128 f16 = 16 MB
static const size_t OFF_BIAS   = OFF_ATTN + (size_t)NROWS*DD*2;      // 4*65536 f32 = 1 MB

// dynamic LDS for attention kernel
static const int SMEM_ATTN = 3*(256*32*2) + 8*16*257*4 + 8*16*264*2 + 2*128*4; // 249344 B

// ---------------- CDNA5 async global->LDS copy (16B per lane) ----------------
// CDNA5 ISA: GLOBAL_LOAD_ASYNC_TO_LDS_B128, GV mode.
// VDST VGPR holds the LDS byte offset (== low 32 bits of a generic shared ptr),
// VADDR holds the 64-bit global address. Tracked with ASYNCcnt.
__device__ __forceinline__ void async_copy_b128(const _Float16* gsrc, _Float16* ldst) {
  unsigned int lds_off = (unsigned int)(unsigned long long)ldst;
  asm volatile("global_load_async_to_lds_b128 %0, %1, off"
               :: "v"(lds_off), "v"(gsrc) : "memory");
}
__device__ __forceinline__ void async_wait0() {
  asm volatile("s_wait_asynccnt 0x0" ::: "memory");
}

// ---------------- pack helpers (CDNA5 WMMA 16-bit layouts) ----------------
// A 16x32 f16: lane<16 holds row=lane&15, halves[0:8]=K0..7, halves[8:16]=K8..15
//              lane>=16: K16..23 / K24..31
__device__ __forceinline__ v16h pack_a(const _Float16* src, int stride, int lane) {
  int row = lane & 15;
  int b1 = (lane < 16) ? 0 : 16;
  int b2 = (lane < 16) ? 8 : 24;
  const _Float16* p = src + row * stride;
  v16h a;
#pragma unroll
  for (int t = 0; t < 8; ++t) { a[t] = p[b1 + t]; a[8 + t] = p[b2 + t]; }
  return a;
}
// B 32x16 f16 from an N-major (transposed) source: lane holds column N=lane&15,
// halves[0:16] = K0..15 (lane<16) or K16..31 (lane>=16), contiguous.
__device__ __forceinline__ v16h pack_b(const _Float16* srcT, int stride, int lane) {
  int col = lane & 15;
  int kb = (lane < 16) ? 0 : 16;
  const _Float16* p = srcT + col * stride + kb;
  v16h b;
#pragma unroll
  for (int t = 0; t < 16; ++t) b[t] = p[t];
  return b;
}

// ---------------- weight prep: f32 -> f16, transposed to N-major ----------------
__global__ void prep_weights(const float* __restrict__ Wqkv, const float* __restrict__ Wb,
                             const float* __restrict__ Wg, const float* __restrict__ Wout,
                             _Float16* __restrict__ wcat, _Float16* __restrict__ woutT) {
  int r = blockIdx.x, d = threadIdx.x;
  if (r < 384)      wcat[r * DD + d] = (_Float16)Wqkv[d * 384 + r];
  else if (r < 512) wcat[r * DD + d] = (_Float16)Wg[d * DD + (r - 384)];
  else if (r < 516) wcat[r * DD + d] = (_Float16)Wb[d * 4 + (r - 512)];
  else if (r < 528) wcat[r * DD + d] = (_Float16)0.0f;
  else              woutT[(r - 528) * DD + d] = (_Float16)Wout[d * DD + (r - 528)];
}

// ---------------- LayerNorm over d=128, one row per block ----------------
__global__ void ln_kernel(const float* __restrict__ edges, const float* __restrict__ g,
                          const float* __restrict__ b, _Float16* __restrict__ out) {
  __shared__ float red[DD];
  int row = blockIdx.x, t = threadIdx.x;
  float x = edges[(size_t)row * DD + t];
  red[t] = x; __syncthreads();
#pragma unroll
  for (int s = 64; s > 0; s >>= 1) { if (t < s) red[t] += red[t + s]; __syncthreads(); }
  float mu = red[0] * (1.0f / DD);
  __syncthreads();
  float dx = x - mu;
  red[t] = dx * dx; __syncthreads();
#pragma unroll
  for (int s = 64; s > 0; s >>= 1) { if (t < s) red[t] += red[t + s]; __syncthreads(); }
  float var = red[0] * (1.0f / DD);
  out[(size_t)row * DD + t] = (_Float16)(dx * rsqrtf(var + 1e-5f) * g[t] + b[t]);
}

// ---------------- GEMM1: proj[65536x528] = normed[65536x128] @ Wcat ----------------
__global__ void __launch_bounds__(256) gemm_proj(const _Float16* __restrict__ A,
                                                 const _Float16* __restrict__ Bt,
                                                 _Float16* __restrict__ C) {
  int tile = blockIdx.x * 8 + (threadIdx.x >> 5);
  if (tile >= 4096 * 33) return;
  int mt = tile & 4095;      // consecutive waves share the B tile (L2 reuse)
  int nt = tile >> 12;
  int lane = threadIdx.x & 31;
  int m0 = mt << 4, n0 = nt << 4;
  v8f c = {};
#pragma unroll
  for (int kc = 0; kc < 4; ++kc) {
    v16h a = pack_a(A + (size_t)m0 * DD + kc * 32, DD, lane);
    v16h b = pack_b(Bt + (size_t)n0 * DD + kc * 32, DD, lane);
    c = __builtin_amdgcn_wmma_f32_16x16x32_f16(false, a, false, b, (short)0, c, false, false);
  }
  int col = lane & 15, ro = (lane < 16) ? 0 : 8;
#pragma unroll
  for (int r = 0; r < 8; ++r)
    C[(size_t)(m0 + r + ro) * PLD + n0 + col] = (_Float16)c[r];
}

// ---------------- bias repack: biasbuf[h][j*256+k] (f32) ----------------
__global__ void bias_repack(const _Float16* __restrict__ proj, float* __restrict__ biasbuf) {
  int idx = blockIdx.x * 256 + threadIdx.x;
#pragma unroll
  for (int h = 0; h < NH; ++h)
    biasbuf[h * NROWS + idx] = (float)proj[(size_t)idx * PLD + 512 + h];
}

// ---------------- attention: one block per (h,i) ----------------
__global__ void __launch_bounds__(256, 1) attn_kernel(const _Float16* __restrict__ proj,
                                                      const float* __restrict__ biasbuf,
                                                      const int* __restrict__ mask,
                                                      const float* __restrict__ b_g,
                                                      _Float16* __restrict__ attnout) {
  extern __shared__ char smem[];
  _Float16* qS = (_Float16*)smem;           // [256][32]
  _Float16* kS = qS + 256 * 32;             // [256][32]  (N-major B source)
  _Float16* vT = kS + 256 * 32;             // [32][256]  (N-major B source)
  float*    S  = (float*)(vT + 32 * 256);   // [8 waves][16][257]  f32 scores
  _Float16* P  = (_Float16*)(S + 8 * 16 * 257); // [8 waves][16][264] f16 probs
  float*    rm = (float*)(P + 8 * 16 * 264);    // [8][16] row max
  float*    rs = rm + 128;                      // [8][16] row sumexp

  int h = blockIdx.x & 3;
  int i = blockIdx.x >> 2;
  int tid = threadIdx.x, lane = tid & 31, wave = tid >> 5;

  { // stage q,k via async global->LDS (16B chunks, ASYNCcnt); v transposed via VGPRs
    int j = tid;
    const _Float16* base = proj + (size_t)(i * 256 + j) * PLD + h * DHD;
#pragma unroll
    for (int c = 0; c < 4; ++c) {   // one q/k row = 32 halfs = 4 x b128
      async_copy_b128(base + c * 8,       qS + j * DHD + c * 8);
      async_copy_b128(base + 128 + c * 8, kS + j * DHD + c * 8);
    }
#pragma unroll
    for (int t = 0; t < DHD; ++t)
      vT[t * 256 + j] = base[256 + t];
    async_wait0();
  }
  __syncthreads();

  float*    Sw = S + wave * 16 * 257;
  _Float16* Pw = P + wave * 16 * 264;
  int col = lane & 15, ro = (lane < 16) ? 0 : 8;

  for (int half = 0; half < 2; ++half) {
    int j0 = (wave + half * 8) * 16;
    // ---- scores: S[j][k] = q.k*scale + bias[j][k] ----
    v16h a = pack_a(qS + j0 * DHD, DHD, lane);
    float maxp[8];
#pragma unroll
    for (int r = 0; r < 8; ++r) maxp[r] = -3.0e38f;
    for (int nt = 0; nt < 16; ++nt) {
      v16h b = pack_b(kS + nt * 16 * DHD, DHD, lane);
      v8f c = {};
      c = __builtin_amdgcn_wmma_f32_16x16x32_f16(false, a, false, b, (short)0, c, false, false);
      int k = nt * 16 + col;
#pragma unroll
      for (int r = 0; r < 8; ++r) {
        int j = j0 + r + ro;
        float s = c[r] * SCALEV + biasbuf[h * NROWS + j * 256 + k];
        Sw[(r + ro) * 257 + k] = s;
        maxp[r] = fmaxf(maxp[r], s);
      }
    }
#pragma unroll
    for (int r = 0; r < 8; ++r) {
#pragma unroll
      for (int off = 8; off >= 1; off >>= 1)
        maxp[r] = fmaxf(maxp[r], __shfl_xor(maxp[r], off, 32));
    }
    if (col == 0) {
#pragma unroll
      for (int r = 0; r < 8; ++r) rm[wave * 16 + r + ro] = maxp[r];
    }
    __syncthreads();
    // ---- softmax numerator, f16 probs ----
    {
      int row = lane >> 1;
      int cb = (lane & 1) * 128;
      float m = rm[wave * 16 + row];
      float sum = 0.f;
#pragma unroll 4
      for (int t = 0; t < 128; ++t) {
        float p = __expf(Sw[row * 257 + cb + t] - m);
        sum += p;
        Pw[row * 264 + cb + t] = (_Float16)p;
      }
      sum += __shfl_xor(sum, 1, 32);
      if ((lane & 1) == 0) rs[wave * 16 + row] = sum;
    }
    __syncthreads();
    // ---- O = P @ V (K=256 over 8 chunks, 2 N-tiles of dh) ----
    v8f acc0 = {}, acc1 = {};
#pragma unroll
    for (int kc = 0; kc < 8; ++kc) {
      v16h pa  = pack_a(Pw + kc * 32, 264, lane);
      v16h vb0 = pack_b(vT + kc * 32, 256, lane);
      v16h vb1 = pack_b(vT + 16 * 256 + kc * 32, 256, lane);
      acc0 = __builtin_amdgcn_wmma_f32_16x16x32_f16(false, pa, false, vb0, (short)0, acc0, false, false);
      acc1 = __builtin_amdgcn_wmma_f32_16x16x32_f16(false, pa, false, vb1, (short)0, acc1, false, false);
    }
    // ---- gate with sigmoid(g) (0 where mask row is off) and store ----
#pragma unroll
    for (int r = 0; r < 8; ++r) {
      int row = r + ro;
      int j = j0 + row;
      float inv = 1.0f / rs[wave * 16 + row];
      int valid = mask[i * 256 + j];
      size_t prow = (size_t)(i * 256 + j) * PLD;
#pragma unroll
      for (int nn = 0; nn < 2; ++nn) {
        int dh = nn * 16 + col;
        float o = (nn ? acc1[r] : acc0[r]) * inv;
        float gr = (float)proj[prow + 384 + h * DHD + dh] + b_g[h * DHD + dh];
        float gate = valid ? (1.0f / (1.0f + __expf(-gr))) : 0.0f;
        attnout[(size_t)(i * 256 + j) * DD + h * DHD + dh] = (_Float16)(o * gate);
      }
    }
    __syncthreads();
  }
}

// ---------------- GEMM3: out = attnout @ W_out + b_out (f32 out) ----------------
__global__ void __launch_bounds__(256) gemm_out(const _Float16* __restrict__ A,
                                                const _Float16* __restrict__ Bt,
                                                const float* __restrict__ bias,
                                                float* __restrict__ C) {
  int tile = blockIdx.x * 8 + (threadIdx.x >> 5); // 4096*8 tiles exactly
  int mt = tile & 4095;
  int nt = tile >> 12;
  int lane = threadIdx.x & 31;
  int m0 = mt << 4, n0 = nt << 4;
  v8f c = {};
#pragma unroll
  for (int kc = 0; kc < 4; ++kc) {
    v16h a = pack_a(A + (size_t)m0 * DD + kc * 32, DD, lane);
    v16h b = pack_b(Bt + (size_t)n0 * DD + kc * 32, DD, lane);
    c = __builtin_amdgcn_wmma_f32_16x16x32_f16(false, a, false, b, (short)0, c, false, false);
  }
  int col = lane & 15, ro = (lane < 16) ? 0 : 8;
  float bb = bias[n0 + col];
#pragma unroll
  for (int r = 0; r < 8; ++r)
    C[(size_t)(m0 + r + ro) * DD + n0 + col] = c[r] + bb;
}

extern "C" void kernel_launch(void* const* d_in, const int* in_sizes, int n_in,
                              void* d_out, int out_size, void* d_ws, size_t ws_size,
                              hipStream_t stream) {
  (void)in_sizes; (void)n_in; (void)out_size; (void)ws_size;
  const float* edges = (const float*)d_in[0];
  const int*   mask  = (const int*)d_in[1];
  const float* ln_g  = (const float*)d_in[2];
  const float* ln_b  = (const float*)d_in[3];
  const float* Wqkv  = (const float*)d_in[4];
  const float* Wb    = (const float*)d_in[5];
  const float* Wg    = (const float*)d_in[6];
  const float* bg    = (const float*)d_in[7];
  const float* Wout  = (const float*)d_in[8];
  const float* bout  = (const float*)d_in[9];

  char* ws = (char*)d_ws;
  _Float16* normed  = (_Float16*)(ws + OFF_NORMED);
  _Float16* wcat    = (_Float16*)(ws + OFF_WCAT);
  _Float16* woutT   = (_Float16*)(ws + OFF_WOUT);
  _Float16* proj    = (_Float16*)(ws + OFF_PROJ);
  _Float16* attnout = (_Float16*)(ws + OFF_ATTN);
  float*    biasbuf = (float*)(ws + OFF_BIAS);

  hipFuncSetAttribute((const void*)attn_kernel,
                      hipFuncAttributeMaxDynamicSharedMemorySize, SMEM_ATTN);

  prep_weights<<<656, 128, 0, stream>>>(Wqkv, Wb, Wg, Wout, wcat, woutT);
  ln_kernel<<<NROWS, 128, 0, stream>>>(edges, ln_g, ln_b, normed);
  gemm_proj<<<16896, 256, 0, stream>>>(normed, wcat, proj);
  bias_repack<<<256, 256, 0, stream>>>(proj, biasbuf);
  attn_kernel<<<NH * NN, 256, SMEM_ATTN, stream>>>(proj, biasbuf, mask, bg, attnout);
  gemm_out<<<4096, 256, 0, stream>>>(attnout, woutT, bout, (float*)d_out);
}
